// Model_34256659153262
// MI455X (gfx1250) — matrix operand order repaired and checked by reading, not compiled, not run on hardware
//
#include <hip/hip_runtime.h>
#include <hip/hip_bf16.h>

// ---------------------------------------------------------------------------
// MI455X (gfx1250): bf16 WMMA (f32 accum) pipeline.
//  - All B-matrices staged in LDS transposed (bT[n][k], k-stride 32) so A and
//    B fragment loads are contiguous 16B runs -> ds_load_b128.
//  - Global->LDS staging uses GLOBAL_LOAD_ASYNC_TO_LDS_B128 (ASYNCcnt), no
//    VGPR round trip.
//  - Tail handling by index clamping (no exec-divergent staging).
// ---------------------------------------------------------------------------

#define B_DIM 64
#define S_DIM 512
#define C_DIM 862
#define P_DIM 96
#define M_DIM 512
#define H_DIM 128
#define CTILES 54  // ceil(862/16)

typedef __attribute__((ext_vector_type(16))) __bf16 v16bf;
typedef __attribute__((ext_vector_type(8)))  __bf16 v8bf;
typedef __attribute__((ext_vector_type(8)))  float  v8f;

__device__ __forceinline__ int clampc(int c) { return (c < C_DIM) ? c : (C_DIM - 1); }

__device__ __forceinline__ v8f zero8() {
  v8f z = {0.f, 0.f, 0.f, 0.f, 0.f, 0.f, 0.f, 0.f};
  return z;
}

__device__ __forceinline__ v8f wmma_bf16(v16bf a, v16bf b, v8f c) {
  return __builtin_amdgcn_wmma_f32_16x16x32_bf16(
      /*neg_a=*/false, a, /*neg_b=*/false, b,
      /*c_mod=*/(short)0, c, /*reuse_a=*/false, /*reuse_b=*/false);
}

__device__ __forceinline__ v16bf cat8(v8bf lo, v8bf hi) {
  v16bf f;
#pragma unroll
  for (int i = 0; i < 8; ++i) { f[i] = lo[i]; f[8 + i] = hi[i]; }
  return f;
}

// A-matrix 16x32 bf16 fragment from row-major [m][k] LDS tile (2x ds_load_b128)
__device__ __forceinline__ v16bf load_a_frag(const __bf16* base, int stride, int lane) {
  int m = lane & 15, half = lane >> 4;
  const __bf16* p0 = base + m * stride + half * 8;
  v8bf lo = *reinterpret_cast<const v8bf*>(p0);
  v8bf hi = *reinterpret_cast<const v8bf*>(p0 + 16);
  return cat8(lo, hi);
}

// B-matrix 32x16 bf16 fragment from transposed slab bT[n][k] (2x ds_load_b128)
__device__ __forceinline__ v16bf load_b_fragT(const __bf16* tile_base, int lane) {
  int n = lane & 15, half = lane >> 4;
  const __bf16* p0 = tile_base + n * 32 + half * 8;
  v8bf lo = *reinterpret_cast<const v8bf*>(p0);
  v8bf hi = *reinterpret_cast<const v8bf*>(p0 + 16);
  return cat8(lo, hi);
}

// Async global->LDS 16-byte copy (no VGPR data path; tracked by ASYNCcnt).
// Flat LDS pointers keep the LDS byte address in addr[31:0] (ISA 10.2).
__device__ __forceinline__ void async_copy16(void* lds_dst, const void* gsrc) {
  unsigned l = (unsigned)(uintptr_t)lds_dst;
  unsigned long long g = (unsigned long long)(uintptr_t)gsrc;
  asm volatile("global_load_async_to_lds_b128 %0, %1, off" :: "v"(l), "v"(g) : "memory");
}
__device__ __forceinline__ void async_wait0() {
  asm volatile("s_wait_asynccnt 0x0" ::: "memory");
}

// Stage a 32-wide k-slab of row-major bf16 W[n][k_total] into LDS bT[n][0..32)
// as NROWS*4 async b128 copies (128 threads).
template <int NROWS>
__device__ __forceinline__ void stage_bslabT(__bf16* dst, const __bf16* src,
                                             int row_stride, int ks, int t) {
#pragma unroll
  for (int i = 0; i < NROWS * 4 / 128; ++i) {
    int q = t + i * 128;             // one uint4 (8 bf16) per copy
    int n = q >> 2, j = q & 3;
    async_copy16(dst + (size_t)q * 8, src + (size_t)n * row_stride + ks + j * 8);
  }
}

// --------------------------- K0: weight prep --------------------------------
__global__ void prep_weights(const float* __restrict__ W1, const float* __restrict__ Wp,
                             const float* __restrict__ W2, const float* __restrict__ memv,
                             __bf16* w1b, __bf16* wpb, __bf16* memb, __bf16* memT,
                             __bf16* w2b) {
  int i = blockIdx.x * blockDim.x + threadIdx.x;
  if (i < H_DIM * S_DIM) w1b[i] = (__bf16)W1[i];        // [h][s] row-major
  if (i < H_DIM * H_DIM) wpb[i] = (__bf16)Wp[i];        // [g][h] row-major
  if (i < M_DIM * H_DIM) {
    memb[i] = (__bf16)memv[i];                          // [f][d] row-major
    int d = i >> 9, f = i & 511;                        // memT[d][f] = memory[f][d]
    memT[i] = (__bf16)memv[f * H_DIM + d];
  }
  if (i < P_DIM * 2 * H_DIM) w2b[i] = (__bf16)W2[i];    // [p][2H] row-major
}

// --------------------------- K1: GEMM1 + sigmoid ----------------------------
// r1[b*C+c][h] = sigmoid( sum_s x[b,s,c] * W1[h,s] + b1[h] )
__global__ void gemm1_sigmoid(const float* __restrict__ x, const __bf16* __restrict__ w1b,
                              const float* __restrict__ b1, __bf16* __restrict__ r1) {
  __shared__ __align__(16) __bf16 a_lds[16 * 32];
  __shared__ __align__(16) __bf16 b_lds[128 * 32];   // bT[h][k]
  const int t = threadIdx.x, lane = t & 31, w = t >> 5;
  const int c0 = blockIdx.x * 16, b = blockIdx.y;
  v8f acc[2] = {zero8(), zero8()};

  // A staging: thread t owns (m = t&15, kbase = t>>4), 4 elems k = kbase+8i.
  // Clamped c => no divergence; base pointer advanced per k-step so the 4
  // loads become global_load_b32 with immediate offsets.
  const int am = t & 15, akb = t >> 4;
  const int ac = clampc(c0 + am);
  const float* px = x + (size_t)b * S_DIM * C_DIM + (size_t)akb * C_DIM + ac;
  __bf16* pa = a_lds + am * 32 + akb;

  for (int ks = 0; ks < S_DIM; ks += 32) {
#pragma unroll
    for (int i = 0; i < 4; ++i)
      pa[i * 8] = (__bf16)px[(size_t)(i * 8) * C_DIM];
    px += (size_t)32 * C_DIM;
    stage_bslabT<128>(b_lds, w1b, S_DIM, ks, t);     // bT[h][k] = W1[h][ks+k]
    async_wait0();
    __syncthreads();
    v16bf af = load_a_frag(a_lds, 32, lane);
#pragma unroll
    for (int tile = 0; tile < 2; ++tile) {
      v16bf bfr = load_b_fragT(b_lds + (w * 32 + tile * 16) * 32, lane);
      acc[tile] = wmma_bf16(af, bfr, acc[tile]);
    }
    __syncthreads();
  }
  const int n = lane & 15, half = lane >> 4;
#pragma unroll
  for (int tile = 0; tile < 2; ++tile) {
    int h = w * 32 + tile * 16 + n;
    float bias = b1[h];
#pragma unroll
    for (int v = 0; v < 8; ++v) {
      int c = c0 + v + 8 * half;
      if (c < C_DIM) {
        float val = acc[tile][v] + bias;
        val = 1.f / (1.f + __expf(-val));
        r1[((size_t)b * C_DIM + c) * H_DIM + h] = (__bf16)val;
      }
    }
  }
}

// --------------------------- K2: projector GEMM -----------------------------
// r[n][g] = sum_h r1[n][h] * Wp[g][h] + bp[g]
__global__ void gemm2_proj(const __bf16* __restrict__ r1, const __bf16* __restrict__ wpb,
                           const float* __restrict__ bp, __bf16* __restrict__ r) {
  __shared__ __align__(16) __bf16 a_lds[16 * 32];
  __shared__ __align__(16) __bf16 b_lds[128 * 32];   // bT[g][k]
  const int t = threadIdx.x, lane = t & 31, w = t >> 5;
  const int c0 = blockIdx.x * 16, b = blockIdx.y;
  v8f acc[2] = {zero8(), zero8()};

  for (int ks = 0; ks < H_DIM; ks += 32) {
    if (t < 64) {                    // A: 64 async b128 copies (clamped rows)
      int m = t >> 2, k4 = t & 3;
      int cc = clampc(c0 + m);
      async_copy16(a_lds + (size_t)t * 8,
                   r1 + ((size_t)b * C_DIM + cc) * H_DIM + ks + k4 * 8);
    }
    stage_bslabT<128>(b_lds, wpb, H_DIM, ks, t);     // bT[g][k] = Wp[g][ks+k]
    async_wait0();
    __syncthreads();
    v16bf af = load_a_frag(a_lds, 32, lane);
#pragma unroll
    for (int tile = 0; tile < 2; ++tile) {
      v16bf bfr = load_b_fragT(b_lds + (w * 32 + tile * 16) * 32, lane);
      acc[tile] = wmma_bf16(af, bfr, acc[tile]);
    }
    __syncthreads();
  }
  const int n = lane & 15, half = lane >> 4;
#pragma unroll
  for (int tile = 0; tile < 2; ++tile) {
    int g = w * 32 + tile * 16 + n;
    float bias = bp[g];
#pragma unroll
    for (int v = 0; v < 8; ++v) {
      int c = c0 + v + 8 * half;
      if (c < C_DIM) {
        float val = acc[tile][v] + bias;
        r[((size_t)b * C_DIM + c) * H_DIM + g] = (__bf16)val;
      }
    }
  }
}

// ------------------- K3: fused attention + predict + transpose --------------
__global__ void attn_predict(const __bf16* __restrict__ r, const __bf16* __restrict__ memb,
                             const __bf16* __restrict__ memT, const __bf16* __restrict__ w2b,
                             const float* __restrict__ b2, float* __restrict__ out) {
  // 32KB buffer time-multiplexed: B-slab staging (max 512x32 bf16) <-> f32 scores
  __shared__ __align__(16) char smem[16 * 512 * 4];
  __shared__ __align__(16) __bf16 att[16 * 512];
  __shared__ __align__(16) __bf16 r_tile[16 * 128];
  __shared__ __align__(16) __bf16 out_t[16 * 128];
  __shared__ float redp[16 * 8];
  __shared__ float rowmax[16];
  __shared__ float rowsum[16];
  float*  scores = reinterpret_cast<float*>(smem);
  __bf16* bslab  = reinterpret_cast<__bf16*>(smem);

  const int t = threadIdx.x, lane = t & 31, w = t >> 5;
  const int c0 = blockIdx.x * 16, b = blockIdx.y;

  // Phase 1: stage r_tile via async copies (clamped rows)
#pragma unroll
  for (int i = 0; i < 2; ++i) {
    int q = t + i * 128;             // 256 uint4 = 2048 bf16
    int m = q >> 4, k4 = q & 15;
    int cc = clampc(c0 + m);
    async_copy16(r_tile + (size_t)q * 8,
                 r + ((size_t)b * C_DIM + cc) * H_DIM + k4 * 8);
  }
  async_wait0();
  __syncthreads();

  // Phase 2: scores GEMM — wave w owns codebook cols [w*128, w*128+128)
  v8f sacc[8];
#pragma unroll
  for (int j = 0; j < 8; ++j) sacc[j] = zero8();
  for (int ks = 0; ks < H_DIM; ks += 32) {
    stage_bslabT<512>(bslab, memb, H_DIM, ks, t);    // bT[f][k] = memory[f][ks+k]
    async_wait0();
    __syncthreads();
    v16bf af = load_a_frag(r_tile + ks, 128, lane);
#pragma unroll
    for (int j2 = 0; j2 < 8; ++j2) {
      v16bf bfr = load_b_fragT(bslab + (w * 128 + j2 * 16) * 32, lane);
      sacc[j2] = wmma_bf16(af, bfr, sacc[j2]);
    }
    __syncthreads();
  }
  // Phase 3: spill scores (f32) to LDS
  {
    int n = lane & 15, half = lane >> 4;
#pragma unroll
    for (int j2 = 0; j2 < 8; ++j2)
#pragma unroll
      for (int v = 0; v < 8; ++v)
        scores[(v + 8 * half) * 512 + w * 128 + j2 * 16 + n] = sacc[j2][v];
  }
  __syncthreads();

  // Phase 4: softmax over 512 per row (8 threads/row, 64 cols each)
  {
    int m = t >> 3, seg = t & 7;
    float mx = -3.4e38f;
    for (int f = seg * 64; f < seg * 64 + 64; ++f) mx = fmaxf(mx, scores[m * 512 + f]);
    redp[m * 8 + seg] = mx;
    __syncthreads();
    if (seg == 0) {
      float v = redp[m * 8];
      for (int q = 1; q < 8; ++q) v = fmaxf(v, redp[m * 8 + q]);
      rowmax[m] = v;
    }
    __syncthreads();
    float rm = rowmax[m];
    float sm = 0.f;
    for (int f = seg * 64; f < seg * 64 + 64; ++f) sm += __expf(scores[m * 512 + f] - rm);
    redp[m * 8 + seg] = sm;
    __syncthreads();
    if (seg == 0) {
      float v = 0.f;
      for (int q = 0; q < 8; ++q) v += redp[m * 8 + q];
      rowsum[m] = v;
    }
    __syncthreads();
    float inv = 1.f / rowsum[m];
    for (int f = seg * 64; f < seg * 64 + 64; ++f)
      att[m * 512 + f] = (__bf16)(__expf(scores[m * 512 + f] - rm) * inv);
  }
  __syncthreads();

  // Phase 5: out GEMM — out[16,128] = att @ mem, wave w owns cols [w*32,+32)
  v8f oacc[2] = {zero8(), zero8()};
  for (int ks = 0; ks < M_DIM; ks += 32) {
    stage_bslabT<128>(bslab, memT, M_DIM, ks, t);    // bT[d][k] = memT[d][ks+k]
    async_wait0();
    __syncthreads();
    v16bf af = load_a_frag(att + ks, 512, lane);
#pragma unroll
    for (int tile = 0; tile < 2; ++tile) {
      v16bf bfr = load_b_fragT(bslab + (w * 32 + tile * 16) * 32, lane);
      oacc[tile] = wmma_bf16(af, bfr, oacc[tile]);
    }
    __syncthreads();
  }
  // Phase 6: spill out (bf16) — head of concat for predict
  {
    int n = lane & 15, half = lane >> 4;
#pragma unroll
    for (int tile = 0; tile < 2; ++tile)
#pragma unroll
      for (int v = 0; v < 8; ++v)
        out_t[(v + 8 * half) * 128 + w * 32 + tile * 16 + n] = (__bf16)oacc[tile][v];
  }
  __syncthreads();

  // Phase 7: predict GEMM — res[16,96] = [out_t | r_tile] @ W2^T, K=256
  v8f pacc[2] = {zero8(), zero8()};
  for (int ks = 0; ks < 2 * H_DIM; ks += 32) {
    stage_bslabT<96>(bslab, w2b, 2 * H_DIM, ks, t);  // bT[p][k] = W2[p][ks+k]
    async_wait0();
    __syncthreads();
    const __bf16* asrc = (ks < H_DIM) ? (out_t + ks) : (r_tile + (ks - H_DIM));
    v16bf af = load_a_frag(asrc, 128, lane);
    int idx = 0;
    for (int tt = w; tt < 6; tt += 4, ++idx) {       // wave-uniform trip counts
      v16bf bfr = load_b_fragT(bslab + tt * 16 * 32, lane);
      pacc[idx] = wmma_bf16(af, bfr, pacc[idx]);
    }
    __syncthreads();
  }
  // Phase 8: bias + transposed store -> out[b, p, c]
  {
    int n = lane & 15, half = lane >> 4;
    int idx = 0;
    for (int tt = w; tt < 6; tt += 4, ++idx) {
      int p = tt * 16 + n;
      float bias = b2[p];
#pragma unroll
      for (int v = 0; v < 8; ++v) {
        int c = c0 + v + 8 * half;
        if (c < C_DIM)
          out[((size_t)b * P_DIM + p) * C_DIM + c] = pacc[idx][v] + bias;
      }
    }
  }
}

// ---------------------------------------------------------------------------
extern "C" void kernel_launch(void* const* d_in, const int* in_sizes, int n_in,
                              void* d_out, int out_size, void* d_ws, size_t ws_size,
                              hipStream_t stream) {
  const float* x    = (const float*)d_in[0];
  const float* W1   = (const float*)d_in[1];
  const float* b1   = (const float*)d_in[2];
  const float* Wp   = (const float*)d_in[3];
  const float* bp   = (const float*)d_in[4];
  const float* W2   = (const float*)d_in[5];
  const float* b2   = (const float*)d_in[6];
  const float* memv = (const float*)d_in[7];
  float* res = (float*)d_out;

  char* ws = (char*)d_ws;
  __bf16* w1b  = (__bf16*)(ws + 0);         // 128x512 bf16 = 131072 B
  __bf16* wpb  = (__bf16*)(ws + 131072);    // 128x128 bf16 =  32768 B
  __bf16* memb = (__bf16*)(ws + 163840);    // 512x128 bf16 = 131072 B
  __bf16* memT = (__bf16*)(ws + 294912);    // 128x512 bf16 = 131072 B
  __bf16* w2b  = (__bf16*)(ws + 425984);    //  96x256 bf16 =  49152 B
  __bf16* r1   = (__bf16*)(ws + 475136);    // 55168x128 bf16 = 14123008 B
  __bf16* r    = (__bf16*)(ws + 475136 + 14123008);

  prep_weights<<<256, 256, 0, stream>>>(W1, Wp, W2, memv, w1b, wpb, memb, memT, w2b);
  gemm1_sigmoid<<<dim3(CTILES, B_DIM), 128, 0, stream>>>(x, w1b, b1, r1);
  gemm2_proj   <<<dim3(CTILES, B_DIM), 128, 0, stream>>>(r1, wpb, bp, r);
  attn_predict <<<dim3(CTILES, B_DIM), 128, 0, stream>>>(r, memb, memT, w2b, b2, res);
}
